// SA_Module_83726092468885
// MI455X (gfx1250) — matrix operand order repaired and checked by reading, not compiled, not run on hardware
//
#include <hip/hip_runtime.h>
#include <hip/hip_bf16.h>

// MI455X / gfx1250 flash self-attention (SAGAN block), wave32 + WMMA f16->f32.
// Split-j flash attention: 2 waves per 16-row query block, DPP softmax reductions.

typedef __attribute__((ext_vector_type(16))) _Float16 v16h;
typedef __attribute__((ext_vector_type(8)))  float    v8f;
typedef __attribute__((ext_vector_type(4)))  float    f32x4;

#define B_DIM 4
#define C_DIM 128
#define N_DIM 4096

// ---------------------------------------------------------------------------
// DPP lane-permute (VALU, no LDS traffic). CTRL: 0xB1 quad_perm(1,0,3,2)=xor1,
// 0x4E quad_perm(2,3,0,1)=xor2, 0x141 row_half_mirror, 0x140 row_mirror.
template <int CTRL>
__device__ __forceinline__ float dpp_f(float v) {
  union { float f; int i; } u, r;
  u.f = v;
  r.i = __builtin_amdgcn_update_dpp(u.i, u.i, CTRL, 0xF, 0xF, true);
  return r.f;
}
// max/sum across the 16-lane half (stays within a DPP row of 16).
__device__ __forceinline__ float rowmax16(float t) {
  t = fmaxf(t, dpp_f<0xB1>(t));
  t = fmaxf(t, dpp_f<0x4E>(t));
  t = fmaxf(t, dpp_f<0x141>(t));
  t = fmaxf(t, dpp_f<0x140>(t));
  return t;
}
__device__ __forceinline__ float rowsum16(float t) {
  t += dpp_f<0xB1>(t);
  t += dpp_f<0x4E>(t);
  t += dpp_f<0x141>(t);
  t += dpp_f<0x140>(t);
  return t;
}

// ---------------------------------------------------------------------------
// A-tile loader: 16x32 f16, row-major source with leading dim `ld`.
__device__ inline v16h load_a_tile(const _Float16* base, int ld, int lane) {
  const int r  = lane & 15;
  const int hi = lane >> 4;
  const _Float16* p = base + (size_t)r * ld + hi * 8;
  union { v16h v; f32x4 q[2]; } u;
  u.q[0] = *(const f32x4*)(p);
  u.q[1] = *(const f32x4*)(p + 16);
  return u.v;
}
// B-tile loader: 32x16 f16 (KxN); column n = 32 K-contiguous halves, stride ld.
__device__ inline v16h load_b_tile(const _Float16* base, int ld, int lane) {
  const int c  = lane & 15;
  const int hi = lane >> 4;
  const _Float16* p = base + (size_t)c * ld + hi * 8;
  union { v16h v; f32x4 q[2]; } u;
  u.q[0] = *(const f32x4*)(p);
  u.q[1] = *(const f32x4*)(p + 16);
  return u.v;
}

// ---------------------------------------------------------------------------
// 1x1-conv projection: y[d,n] = sum_c W[d,c] x[b,c,n] + bias[d], emitted as f16.
__global__ __launch_bounds__(256) void qkv_proj_kernel(
    const float* __restrict__ x, const float* __restrict__ W,
    const float* __restrict__ bias, _Float16* __restrict__ out, int trans)
{
  __shared__ __align__(16) float wS[C_DIM * C_DIM];   // 64 KB
  const int t  = threadIdx.x;
  const int b  = blockIdx.x >> 5;
  const int n0 = (blockIdx.x & 31) * 128;

  for (int i = t; i < C_DIM * C_DIM; i += 256) wS[i] = W[i];
  __syncthreads();

  const int p  = t & 127;
  const int d0 = (t >> 7) * 64;
  const float* xcol = x + (size_t)b * C_DIM * N_DIM + n0 + p;

  float acc[64];
  #pragma unroll
  for (int i = 0; i < 64; ++i) acc[i] = 0.f;

  for (int c = 0; c < C_DIM; c += 8) {
    float xr[8];
    #pragma unroll
    for (int cc = 0; cc < 8; ++cc) xr[cc] = xcol[(size_t)(c + cc) * N_DIM];
    #pragma unroll
    for (int dd = 0; dd < 64; ++dd) {
      const f32x4* wrow = (const f32x4*)&wS[(d0 + dd) * C_DIM + c];
      f32x4 w0 = wrow[0], w1 = wrow[1];
      acc[dd] = fmaf(w0[0], xr[0], acc[dd]);
      acc[dd] = fmaf(w0[1], xr[1], acc[dd]);
      acc[dd] = fmaf(w0[2], xr[2], acc[dd]);
      acc[dd] = fmaf(w0[3], xr[3], acc[dd]);
      acc[dd] = fmaf(w1[0], xr[4], acc[dd]);
      acc[dd] = fmaf(w1[1], xr[5], acc[dd]);
      acc[dd] = fmaf(w1[2], xr[6], acc[dd]);
      acc[dd] = fmaf(w1[3], xr[7], acc[dd]);
    }
  }

  if (trans) {
    _Float16* o = out + ((size_t)b * N_DIM + n0 + p) * C_DIM + d0;
    #pragma unroll
    for (int dd = 0; dd < 64; ++dd) o[dd] = (_Float16)(acc[dd] + bias[d0 + dd]);
  } else {
    _Float16* o = out + ((size_t)b * C_DIM + d0) * N_DIM + n0 + p;
    #pragma unroll
    for (int dd = 0; dd < 64; ++dd) o[(size_t)dd * N_DIM] = (_Float16)(acc[dd] + bias[d0 + dd]);
  }
}

// ---------------------------------------------------------------------------
// Flash attention, split-j: block = 4 waves = 2 row-blocks x 2 j-slices.
__global__ __launch_bounds__(128) void flash_attn_kernel(
    const _Float16* __restrict__ Qh,  // [B][N][C]
    const _Float16* __restrict__ Kh,  // [B][N][C]
    const _Float16* __restrict__ Vh,  // [B][C][N]
    const float* __restrict__ x,      // [B][C][N]
    const float* __restrict__ gammap,
    float* __restrict__ out)          // [B][C][N]
{
  __shared__ __align__(16) _Float16 pS[4][16 * 32];      // per-wave P transpose
  __shared__ __align__(16) float    oS[4][C_DIM * 16];   // per-wave raw O (8 KB each)
  __shared__ float mS[4][16];
  __shared__ float lS[4][16];

  const int lane = threadIdx.x & 31;
  const int w    = threadIdx.x >> 5;
  const int s    = w & 1;                    // j-slice
  const int rbk  = blockIdx.x * 2 + (w >> 1);// 1024 row-blocks total
  const int b    = rbk >> 8;                 // 256 row-blocks per batch
  const int i0   = (rbk & 255) * 16;
  const float gamma = gammap[0];

  const _Float16* Qb = Qh + (size_t)b * N_DIM * C_DIM;
  const _Float16* Kb = Kh + (size_t)b * N_DIM * C_DIM;
  const _Float16* Vb = Vh + (size_t)b * C_DIM * N_DIM;

  // Resident Q tile: 16 rows x 128 channels = 4 A-tiles
  v16h aq[4];
  #pragma unroll
  for (int cb = 0; cb < 4; ++cb)
    aq[cb] = load_a_tile(Qb + (size_t)i0 * C_DIM + cb * 32, C_DIM, lane);

  v8f O[8];
  {
    v8f z = {};
    #pragma unroll
    for (int co = 0; co < 8; ++co) O[co] = z;
  }
  float m_run[8], l_run[8];
  #pragma unroll
  for (int r = 0; r < 8; ++r) { m_run[r] = -1e30f; l_run[r] = 0.f; }

  _Float16* pW = pS[w];
  const int hiL = lane >> 4, loL = lane & 15;

  for (int chunk = s; chunk < N_DIM / 32; chunk += 2) {
    const int j0 = chunk * 32;
    // ---- S = Q Kt for 16i x 32j (two 16x16 C/D tiles, K=C in 4 chunks) ----
    v8f s0 = {}, s1 = {};
    #pragma unroll
    for (int cb = 0; cb < 4; ++cb) {
      v16h bk = load_b_tile(Kb + (size_t)j0 * C_DIM + cb * 32, C_DIM, lane);
      s0 = __builtin_amdgcn_wmma_f32_16x16x32_f16(false, aq[cb], false, bk,
                                                  (short)0, s0, false, false);
    }
    #pragma unroll
    for (int cb = 0; cb < 4; ++cb) {
      v16h bk = load_b_tile(Kb + (size_t)(j0 + 16) * C_DIM + cb * 32, C_DIM, lane);
      s1 = __builtin_amdgcn_wmma_f32_16x16x32_f16(false, aq[cb], false, bk,
                                                  (short)0, s1, false, false);
    }

    // ---- online softmax; rows at M = r + 8*(lane>=16), j = lane&15 ----
    float alpha[8];
    #pragma unroll
    for (int r = 0; r < 8; ++r) {
      const float t  = rowmax16(fmaxf(s0[r], s1[r]));
      const float nm = fmaxf(m_run[r], t);
      alpha[r] = __expf(m_run[r] - nm);
      m_run[r] = nm;
      const float p0 = __expf(s0[r] - nm);
      const float p1 = __expf(s1[r] - nm);
      s0[r] = p0; s1[r] = p1;
      l_run[r] = l_run[r] * alpha[r] + rowsum16(p0 + p1);
    }
    #pragma unroll
    for (int co = 0; co < 8; ++co)
      #pragma unroll
      for (int r = 0; r < 8; ++r) O[co][r] *= alpha[r];

    // ---- transpose P (C/D layout -> A layout) through per-wave LDS ----
    #pragma unroll
    for (int r = 0; r < 8; ++r) {
      pW[(r + 8 * hiL) * 32 + loL]      = (_Float16)s0[r];
      pW[(r + 8 * hiL) * 32 + 16 + loL] = (_Float16)s1[r];
    }
    asm volatile("s_wait_dscnt 0" ::: "memory");
    v16h ap = load_a_tile(pW, 32, lane);

    // ---- O(16i x 128c) += P(16x32) * Vt(32j x 16c) ----
    #pragma unroll
    for (int co = 0; co < 8; ++co) {
      v16h bv = load_b_tile(Vb + (size_t)(co * 16) * N_DIM + j0, N_DIM, lane);
      O[co] = __builtin_amdgcn_wmma_f32_16x16x32_f16(false, ap, false, bv,
                                                     (short)0, O[co], false, false);
    }
  }

  // ---- publish raw partials (O, m, l) to LDS ----
  float* oW = oS[w];
  #pragma unroll
  for (int co = 0; co < 8; ++co)
    #pragma unroll
    for (int r = 0; r < 8; ++r)
      oW[(co * 16 + loL) * 16 + (r + 8 * hiL)] = O[co][r];
  if (loL == 0) {
    #pragma unroll
    for (int r = 0; r < 8; ++r) {
      mS[w][8 * hiL + r] = m_run[r];
      lS[w][8 * hiL + r] = l_run[r];
    }
  }
  __syncthreads();

  // ---- even wave of each pair merges the two j-slices (log-sum-exp) ----
  if ((w & 1) == 0) {
    const float* o0 = oS[w];
    const float* o1 = oS[w + 1];
    f32x4 F0[4], F1[4], IL[4];
    #pragma unroll
    for (int q = 0; q < 4; ++q) {
      #pragma unroll
      for (int e = 0; e < 4; ++e) {
        const int i = q * 4 + e;
        const float m0 = mS[w][i], m1 = mS[w + 1][i];
        const float M  = fmaxf(m0, m1);
        const float e0 = __expf(m0 - M);
        const float e1 = __expf(m1 - M);
        const float L  = lS[w][i] * e0 + lS[w + 1][i] * e1;
        F0[q][e] = e0;
        F1[q][e] = e1;
        IL[q][e] = gamma / L;
      }
    }
    #pragma unroll
    for (int k = 0; k < 4; ++k) {
      const int c = k * 32 + lane;
      const float* p0 = o0 + c * 16;
      const float* p1 = o1 + c * 16;
      const size_t base = ((size_t)b * C_DIM + c) * N_DIM + i0;
      #pragma unroll
      for (int q = 0; q < 4; ++q) {
        f32x4 a  = *(const f32x4*)(p0 + q * 4);
        f32x4 bb = *(const f32x4*)(p1 + q * 4);
        f32x4 xv = *(const f32x4*)(x + base + q * 4);
        *(f32x4*)(out + base + q * 4) = (a * F0[q] + bb * F1[q]) * IL[q] + xv;
      }
    }
  }
}

// ---------------------------------------------------------------------------
extern "C" void kernel_launch(void* const* d_in, const int* in_sizes, int n_in,
                              void* d_out, int out_size, void* d_ws, size_t ws_size,
                              hipStream_t stream) {
  const float* x     = (const float*)d_in[0];
  const float* Wq    = (const float*)d_in[1];
  const float* bq    = (const float*)d_in[2];
  const float* Wk    = (const float*)d_in[3];
  const float* bk    = (const float*)d_in[4];
  const float* Wv    = (const float*)d_in[5];
  const float* bv    = (const float*)d_in[6];
  const float* gamma = (const float*)d_in[7];
  float* out = (float*)d_out;

  const size_t elems = (size_t)B_DIM * N_DIM * C_DIM;   // 2M halves = 4 MB each
  _Float16* Qh = (_Float16*)d_ws;
  _Float16* Kh = Qh + elems;
  _Float16* Vh = Kh + elems;

  dim3 gp(B_DIM * (N_DIM / 128));   // 128 blocks
  qkv_proj_kernel<<<gp, 256, 0, stream>>>(x, Wq, bq, Qh, 1);
  qkv_proj_kernel<<<gp, 256, 0, stream>>>(x, Wk, bk, Kh, 1);
  qkv_proj_kernel<<<gp, 256, 0, stream>>>(x, Wv, bv, Vh, 0);

  // 1024 row-blocks x 2 j-slices / 4 waves per block = 512 blocks
  flash_attn_kernel<<<dim3(B_DIM * (N_DIM / 16) / 2), 128, 0, stream>>>(
      Qh, Kh, Vh, x, gamma, out);
}
